// MultiHeadedAttention_67963562492598
// MI455X (gfx1250) — matrix operand order repaired and checked by reading, not compiled, not run on hardware
//
#include <hip/hip_runtime.h>
#include <hip/hip_bf16.h>
#include <stdint.h>

#define S 192
#define H 8
#define DK 64
#define DM 512

typedef __attribute__((ext_vector_type(16))) __bf16 v16bf;
typedef __attribute__((ext_vector_type(8)))  float  v8f;
typedef __attribute__((ext_vector_type(2)))  __bf16 bf16x2;

union AB {
  v16bf    v;
  uint32_t u[8];
  uint4    q[2];
};

// f32 -> bf16 via native convert (RNE); lowers to v_cvt_pk_bf16_f32
__device__ __forceinline__ uint16_t f2bf(float f) {
  __bf16 h = (__bf16)f;
  return __builtin_bit_cast(uint16_t, h);
}
__device__ __forceinline__ uint32_t pk_cvt(float lo, float hi) {
  bf16x2 t = { (__bf16)lo, (__bf16)hi };
  return __builtin_bit_cast(uint32_t, t);
}
__device__ __forceinline__ float bf2f(uint16_t b) {
  return (float)__builtin_bit_cast(__bf16, b);
}
// packed bf16 multiply
__device__ __forceinline__ uint32_t mulbf2(uint32_t a, uint32_t b) {
  bf16x2 av = __builtin_bit_cast(bf16x2, a);
  bf16x2 bv = __builtin_bit_cast(bf16x2, b);
  bf16x2 r  = av * bv;
  return __builtin_bit_cast(uint32_t, r);
}
__device__ __forceinline__ void pack_run8(AB& d, int j0, float4 f0, float4 f1) {
  d.u[j0 + 0] = pk_cvt(f0.x, f0.y);
  d.u[j0 + 1] = pk_cvt(f0.z, f0.w);
  d.u[j0 + 2] = pk_cvt(f1.x, f1.y);
  d.u[j0 + 3] = pk_cvt(f1.z, f1.w);
}

// CDNA5 async global->LDS (ASYNCcnt-tracked, bypasses VGPRs). 16 B per lane.
__device__ __forceinline__ void async_g2l_b128(uint32_t lds_off, const void* g) {
  asm volatile("global_load_async_to_lds_b128 %0, %1, off"
               :: "v"(lds_off), "v"((uint64_t)(uintptr_t)g)
               : "memory");
}
__device__ __forceinline__ void wait_async0() {
  asm volatile("s_wait_asynccnt 0x0" ::: "memory");
}

// one 16x16x32 bf16 WMMA step from f32 sources (A: two runs of 8, B: two runs of 8)
__device__ __forceinline__ v8f wmma_f32src(const float* pa, const float* pb, v8f acc) {
  AB a, b;
  pack_run8(a, 0, *(const float4*)pa,        *(const float4*)(pa + 4));
  pack_run8(a, 4, *(const float4*)(pa + 16), *(const float4*)(pa + 20));
  pack_run8(b, 0, *(const float4*)pb,        *(const float4*)(pb + 4));
  pack_run8(b, 4, *(const float4*)(pb + 16), *(const float4*)(pb + 20));
  return __builtin_amdgcn_wmma_f32_16x16x32_bf16(
      false, a.v, false, b.v, (short)0, acc, false, false);
}

// ---------------------------------------------------------------------------
// Kernel 1: five input projections  y = x @ W^T + b  -> bf16, per-head layout.
// One wave per 16x16 output tile; K = 512, two independent accumulator chains
// (even/odd 32-chunks) so WMMAs from the two chains hide the 4-coexec hazard.
// z: 0=q (scaled 1/8), 1=k0, 2=k1, 3=v0, 4=v1 (transposed [H][DK][S]).
// ---------------------------------------------------------------------------
__global__ __launch_bounds__(128) void proj_kernel(
    const float* qx, const float* kx, const float* vx,
    const float* Wq, const float* bq,
    const float* Wk0, const float* bk0,
    const float* Wk1, const float* bk1,
    const float* Wv0, const float* bv0,
    const float* Wv1, const float* bv1,
    uint16_t* oq, uint16_t* ok0, uint16_t* ok1,
    uint16_t* ov0, uint16_t* ov1T) {
  const int lane = threadIdx.x & 31, wave = threadIdx.x >> 5;
  const int r = lane & 15, half = lane >> 4;
  const int s0 = blockIdx.x * 16;
  const int n0 = (blockIdx.y * 4 + wave) * 16;
  const int z = blockIdx.z;

  const float* X; const float* W; const float* bias; uint16_t* dst;
  float osc = 1.0f; int trans = 0;
  switch (z) {
    case 0:  X = qx; W = Wq;  bias = bq;  dst = oq;  osc = 0.125f; break;
    case 1:  X = kx; W = Wk0; bias = bk0; dst = ok0; break;
    case 2:  X = kx; W = Wk1; bias = bk1; dst = ok1; break;
    case 3:  X = vx; W = Wv0; bias = bv0; dst = ov0; break;
    default: X = vx; W = Wv1; bias = bv1; dst = ov1T; trans = 1; break;
  }

  v8f acc0 = {0.f,0.f,0.f,0.f,0.f,0.f,0.f,0.f};
  v8f acc1 = {0.f,0.f,0.f,0.f,0.f,0.f,0.f,0.f};
  const float* xrow = X + (s0 + r) * DM;   // A row for this lane
  const float* wrow = W + (n0 + r) * DM;   // B column n -> W row n

#pragma unroll 2
  for (int kb = 0; kb < DM; kb += 64) {
    acc0 = wmma_f32src(xrow + kb      + half * 8, wrow + kb      + half * 8, acc0);
    acc1 = wmma_f32src(xrow + kb + 32 + half * 8, wrow + kb + 32 + half * 8, acc1);
  }

  const float bn = bias[n0 + r];
  const int h = n0 >> 6, dcol = (n0 & 63) + r;
#pragma unroll
  for (int v = 0; v < 8; ++v) {
    const int srow = s0 + v + half * 8;
    const uint16_t y = f2bf((acc0[v] + acc1[v] + bn) * osc);
    if (!trans) dst[(h * S + srow) * DK + dcol] = y;
    else        dst[(h * DK + dcol) * S + srow] = y;
  }
}

// ---------------------------------------------------------------------------
// Kernel 2: fused trilinear attention for one (k-tile, head).
// 4 waves split the l-loop; softmax over m done in-register (wave32 shuffles).
// LDS: k1 [192][64] bf16 | v1T [64][192] bf16 | p per-wave [16][192] bf16 x4
//      | out reduction [16][64] f32   = 77824 bytes dynamic.
// k1/v1T staged via GLOBAL_LOAD_ASYNC_TO_LDS_B128 (ASYNCcnt).
// ---------------------------------------------------------------------------
__global__ __launch_bounds__(128) void trilin_attn_kernel(
    const uint16_t* qg, const uint16_t* k0g, const uint16_t* k1g,
    const uint16_t* v0g, const uint16_t* v1Tg, uint16_t* attn) {
  extern __shared__ char smem[];
  uint16_t* k1s    = (uint16_t*)smem;              // 24576 B @ LDS off 0
  uint16_t* v1Ts   = (uint16_t*)(smem + 24576);    // 24576 B
  uint16_t* pall   = (uint16_t*)(smem + 49152);    // 4 * 6144 B
  float*    outred = (float*)   (smem + 73728);    // 4096 B

  const int tid = threadIdx.x;
  const int lane = tid & 31, wave = tid >> 5;
  const int r = lane & 15, half = lane >> 4;
  const int s0 = blockIdx.x * 16;
  const int h  = blockIdx.y;

  const uint16_t* k0h = k0g + h * S * DK;
  const uint16_t* v0h = v0g + h * S * DK;

  // async stage: k1 (row-major) and v1^T head slices, 16 B per lane per issue
  {
    const char* g1 = (const char*)(k1g + h * S * DK);
    const char* g2 = (const char*)(v1Tg + h * DK * S);
    const uint32_t l1 = (uint32_t)(uintptr_t)k1s;
    const uint32_t l2 = (uint32_t)(uintptr_t)v1Ts;
#pragma unroll
    for (int i = 0; i < 12; ++i) {             // 12 issues x 128 thr x 16 B = 24576 B
      const uint32_t off = (uint32_t)(i * 128 + tid) * 16u;
      async_g2l_b128(l1 + off, g1 + off);
      async_g2l_b128(l2 + off, g2 + off);
    }
    for (int i = tid; i < 16 * 64; i += 128) outred[i] = 0.f;
    wait_async0();
  }
  __syncthreads();

  uint16_t* pbuf = pall + wave * 16 * S;

  // q fragments (A layout, K = d), invariant over l; q is pre-scaled by 1/8
  AB qa[2];
  {
    const uint16_t* qrow = qg + (h * S + s0 + r) * DK;
#pragma unroll
    for (int c = 0; c < 2; ++c) {
      const uint16_t* p0 = qrow + c * 32 + half * 8;
      qa[c].q[0] = *(const uint4*)p0;
      qa[c].q[1] = *(const uint4*)(p0 + 16);
    }
  }

  v8f oacc[4];
#pragma unroll
  for (int i = 0; i < 4; ++i) oacc[i] = (v8f){0.f,0.f,0.f,0.f,0.f,0.f,0.f,0.f};

  for (int l = wave; l < S; l += 4) {
    // A = (q * k0[l]) elementwise in d, packed bf16 (v_pk_mul_bf16 path)
    AB afr[2];
    {
      const uint16_t* k0row = k0h + l * DK;
#pragma unroll
      for (int c = 0; c < 2; ++c) {
        AB kf;
        const uint16_t* p0 = k0row + c * 32 + half * 8;
        kf.q[0] = *(const uint4*)p0;
        kf.q[1] = *(const uint4*)(p0 + 16);
#pragma unroll
        for (int j = 0; j < 8; ++j) afr[c].u[j] = mulbf2(qa[c].u[j], kf.u[j]);
      }
    }

    // GEMM1: scores[16 x 192] = A(16x64) @ k1^T, 12 m-tiles x 2 K-chunks
    v8f st[12];
#pragma unroll
    for (int t = 0; t < 12; ++t) {
      v8f c0 = {0.f,0.f,0.f,0.f,0.f,0.f,0.f,0.f};
#pragma unroll
      for (int c = 0; c < 2; ++c) {
        AB bfr;
        const uint16_t* pb = k1s + (t * 16 + r) * DK + c * 32 + half * 8;
        bfr.q[0] = *(const uint4*)pb;
        bfr.q[1] = *(const uint4*)(pb + 16);
        c0 = __builtin_amdgcn_wmma_f32_16x16x32_bf16(
            false, afr[c].v, false, bfr.v, (short)0, c0, false, false);
      }
      st[t] = c0;
    }

    // softmax over m (last axis), exact: row max + sum via 16-lane xor shuffles
    float rm[8], rs[8];
#pragma unroll
    for (int v = 0; v < 8; ++v) {
      float mx = st[0][v];
#pragma unroll
      for (int t = 1; t < 12; ++t) mx = fmaxf(mx, st[t][v]);
      mx = fmaxf(mx, __shfl_xor(mx, 1, 32));
      mx = fmaxf(mx, __shfl_xor(mx, 2, 32));
      mx = fmaxf(mx, __shfl_xor(mx, 4, 32));
      mx = fmaxf(mx, __shfl_xor(mx, 8, 32));
      rm[v] = mx;
      rs[v] = 0.f;
    }
#pragma unroll
    for (int t = 0; t < 12; ++t)
#pragma unroll
      for (int v = 0; v < 8; ++v) {
        float e = __expf(st[t][v] - rm[v]);
        st[t][v] = e;
        rs[v] += e;
      }
#pragma unroll
    for (int v = 0; v < 8; ++v) {
      float sm = rs[v];
      sm += __shfl_xor(sm, 1, 32);
      sm += __shfl_xor(sm, 2, 32);
      sm += __shfl_xor(sm, 4, 32);
      sm += __shfl_xor(sm, 8, 32);
      rs[v] = 1.0f / sm;
    }

    // write p (bf16) to this wave's LDS tile, row-major [16][192]
#pragma unroll
    for (int t = 0; t < 12; ++t)
#pragma unroll
      for (int v = 0; v < 8; ++v) {
        const int M = v + half * 8;
        pbuf[M * S + t * 16 + r] = f2bf(st[t][v] * rs[v]);
      }

    // GEMM2: w[16 x 64] = p(16x192) @ v1(192x64); B from v1^T in LDS
    AB pa[6];
    {
      const uint16_t* prow = pbuf + r * S;
#pragma unroll
      for (int c = 0; c < 6; ++c) {
        const uint16_t* p0 = prow + c * 32 + half * 8;
        pa[c].q[0] = *(const uint4*)p0;
        pa[c].q[1] = *(const uint4*)(p0 + 16);
      }
    }
    v8f wacc[4];
#pragma unroll
    for (int td = 0; td < 4; ++td) {
      v8f c0 = {0.f,0.f,0.f,0.f,0.f,0.f,0.f,0.f};
#pragma unroll
      for (int c = 0; c < 6; ++c) {
        AB bfr;
        const uint16_t* pb = v1Ts + (td * 16 + r) * S + c * 32 + half * 8;
        bfr.q[0] = *(const uint4*)pb;
        bfr.q[1] = *(const uint4*)(pb + 16);
        c0 = __builtin_amdgcn_wmma_f32_16x16x32_bf16(
            false, pa[c].v, false, bfr.v, (short)0, c0, false, false);
      }
      wacc[td] = c0;
    }

    // out[k,d] += v0[l,d] * w[k,d]  (rank-1 in l)
    const uint16_t* v0row = v0h + l * DK;
#pragma unroll
    for (int td = 0; td < 4; ++td) {
      const float vv = bf2f(v0row[td * 16 + r]);
#pragma unroll
      for (int v = 0; v < 8; ++v) oacc[td][v] += vv * wacc[td][v];
    }
  }

  // cross-wave reduction over l-partitions
#pragma unroll
  for (int td = 0; td < 4; ++td)
#pragma unroll
    for (int v = 0; v < 8; ++v) {
      const int M = v + half * 8;
      atomicAdd(&outred[M * 64 + td * 16 + r], oacc[td][v]);
    }
  __syncthreads();

  // write concatenated-head attention output (bf16) for the final projection
  for (int i = tid; i < 16 * 64; i += 128) {
    const int M = i >> 6, d = i & 63;
    attn[(s0 + M) * DM + h * DK + d] = f2bf(outred[i]);
  }
}

// ---------------------------------------------------------------------------
// Kernel 3: final projection  out = attn @ Wo^T + bo  (f32 result)
// ---------------------------------------------------------------------------
__global__ __launch_bounds__(128) void outproj_kernel(
    const uint16_t* attn, const float* Wo, const float* bo, float* out) {
  const int lane = threadIdx.x & 31, wave = threadIdx.x >> 5;
  const int r = lane & 15, half = lane >> 4;
  const int s0 = blockIdx.x * 16;
  const int n0 = (blockIdx.y * 4 + wave) * 16;

  v8f acc0 = {0.f,0.f,0.f,0.f,0.f,0.f,0.f,0.f};
  v8f acc1 = {0.f,0.f,0.f,0.f,0.f,0.f,0.f,0.f};
  const uint16_t* arow = attn + (s0 + r) * DM;
  const float*    wrow = Wo + (n0 + r) * DM;

#pragma unroll 2
  for (int kb = 0; kb < DM; kb += 64) {
#pragma unroll
    for (int half_k = 0; half_k < 2; ++half_k) {
      AB a, b;
      const uint16_t* pa = arow + kb + half_k * 32 + half * 8;
      a.q[0] = *(const uint4*)pa;
      a.q[1] = *(const uint4*)(pa + 16);
      const float* pb = wrow + kb + half_k * 32 + half * 8;
      pack_run8(b, 0, *(const float4*)pb,        *(const float4*)(pb + 4));
      pack_run8(b, 4, *(const float4*)(pb + 16), *(const float4*)(pb + 20));
      v8f& acc = half_k ? acc1 : acc0;
      acc = __builtin_amdgcn_wmma_f32_16x16x32_bf16(
          false, a.v, false, b.v, (short)0, acc, false, false);
    }
  }

  const float bn = bo[n0 + r];
#pragma unroll
  for (int v = 0; v < 8; ++v)
    out[(s0 + v + half * 8) * DM + n0 + r] = acc0[v] + acc1[v] + bn;
}

// ---------------------------------------------------------------------------
extern "C" void kernel_launch(void* const* d_in, const int* in_sizes, int n_in,
                              void* d_out, int out_size, void* d_ws, size_t ws_size,
                              hipStream_t stream) {
  const float* query = (const float*)d_in[0];
  const float* key   = (const float*)d_in[1];
  const float* value = (const float*)d_in[2];
  const float* Wq  = (const float*)d_in[3];  const float* bq  = (const float*)d_in[4];
  const float* Wk0 = (const float*)d_in[5];  const float* bk0 = (const float*)d_in[6];
  const float* Wk1 = (const float*)d_in[7];  const float* bk1 = (const float*)d_in[8];
  const float* Wv0 = (const float*)d_in[9];  const float* bv0 = (const float*)d_in[10];
  const float* Wv1 = (const float*)d_in[11]; const float* bv1 = (const float*)d_in[12];
  const float* Wo  = (const float*)d_in[13]; const float* bo  = (const float*)d_in[14];

  char* ws = (char*)d_ws;
  const size_t TEN = (size_t)H * S * DK * sizeof(uint16_t);  // 196608 B
  uint16_t* qb   = (uint16_t*)(ws + 0 * TEN);
  uint16_t* k0b  = (uint16_t*)(ws + 1 * TEN);
  uint16_t* k1b  = (uint16_t*)(ws + 2 * TEN);
  uint16_t* v0b  = (uint16_t*)(ws + 3 * TEN);
  uint16_t* v1Tb = (uint16_t*)(ws + 4 * TEN);
  uint16_t* attn = (uint16_t*)(ws + 5 * TEN);                // [S][DM] bf16

  static int lds_cfg = 0;
  if (!lds_cfg) {  // host-side attribute (not a stream op; graph-capture safe)
    hipFuncSetAttribute((const void*)trilin_attn_kernel,
                        hipFuncAttributeMaxDynamicSharedMemorySize, 77824);
    lds_cfg = 1;
  }

  proj_kernel<<<dim3(12, 8, 5), 128, 0, stream>>>(
      query, key, value, Wq, bq, Wk0, bk0, Wk1, bk1, Wv0, bv0, Wv1, bv1,
      qb, k0b, k1b, v0b, v1Tb);

  trilin_attn_kernel<<<dim3(12, 8), 128, 77824, stream>>>(
      qb, k0b, k1b, v0b, v1Tb, attn);

  outproj_kernel<<<dim3(12, 8), 128, 0, stream>>>(attn, Wo, bo, (float*)d_out);
}
